// MultiHeadAttentionXL_2989297238054
// MI455X (gfx1250) — matrix operand order repaired and checked by reading, not compiled, not run on hardware
//
#include <hip/hip_runtime.h>
#include <math.h>

typedef __bf16 bf16;
typedef bf16 v16bf __attribute__((ext_vector_type(16)));
typedef bf16 v8bf  __attribute__((ext_vector_type(8)));
typedef float v8f  __attribute__((ext_vector_type(8)));
typedef unsigned int u32x4 __attribute__((ext_vector_type(4)));
typedef int i32x4 __attribute__((ext_vector_type(4)));
typedef int i32x8 __attribute__((ext_vector_type(8)));

#define CS 1024
#define PS 1024
#define BSZ 4
#define DIN 1024
#define NH 16
#define DH 64
#define KSQ (CS + PS)     // 2048
#define HD (NH * DH)      // 1024

// ---------------------------------------------------------------------------
// Tensor Data Mover: issue a 2D bf16 tile load (global -> LDS).
// D# packing per CDNA5 ISA ch.8: group0 = {count/type/addrs}, group1 = dims.
// OOB rows (Y >= tensor_d1) are zero-filled by the TDM.
// ---------------------------------------------------------------------------
__device__ __forceinline__ void tdm_load_2d(unsigned lds_off, const void* gptr,
                                            unsigned tile_d0, unsigned tile_d1,
                                            unsigned tensor_d0, unsigned tensor_d1,
                                            unsigned long long stride0) {
  unsigned long long ga = (unsigned long long)(uintptr_t)gptr;
  u32x4 g0;
  g0[0] = 1u;                                         // count=1, user D#
  g0[1] = lds_off;                                    // LDS byte address
  g0[2] = (unsigned)(ga & 0xffffffffull);             // global_addr[31:0]
  g0[3] = (unsigned)((ga >> 32) & 0x01ffffffull)      // global_addr[56:32]
          | (2u << 30);                               // type = 2 ("image")
  i32x8 g1;
  g1[0] = (int)(1u << 16);                            // data_size=1 (2B), mask=0
  g1[1] = (int)((tensor_d0 & 0xffffu) << 16);         // tensor_dim0[15:0]
  g1[2] = (int)((tensor_d0 >> 16) | ((tensor_d1 & 0xffffu) << 16));
  g1[3] = (int)((tensor_d1 >> 16) | (tile_d0 << 16)); // tile_dim0
  g1[4] = (int)(tile_d1 & 0xffffu);                   // tile_dim1, tile_dim2=0
  g1[5] = (int)(stride0 & 0xffffffffull);             // tensor_dim0_stride lo
  g1[6] = (int)((stride0 >> 32) & 0xffffull);         // stride hi, dim1_stride=0
  g1[7] = 0;
  i32x4 z4 = {0, 0, 0, 0};
#if defined(__clang_major__) && (__clang_major__ >= 23)
  i32x8 z8 = {0, 0, 0, 0, 0, 0, 0, 0};
  __builtin_amdgcn_tensor_load_to_lds(g0, g1, z4, z4, z8, 0);
#else
  __builtin_amdgcn_tensor_load_to_lds(g0, g1, z4, z4, 0);
#endif
}

// ---------------------------------------------------------------------------
// f32 -> bf16 converts
// ---------------------------------------------------------------------------
__global__ void __launch_bounds__(256) cvt_copy(const float* __restrict__ in,
                                                bf16* __restrict__ out, long n) {
  long i = (long)blockIdx.x * blockDim.x + threadIdx.x;
  long stride = (long)gridDim.x * blockDim.x;
  for (; i < n; i += stride) out[i] = (bf16)in[i];
}

__global__ void __launch_bounds__(256) cvt_transpose(const float* __restrict__ in,
                                                     bf16* __restrict__ out,
                                                     int K, int N) {
  long n = (long)K * N;
  long i = (long)blockIdx.x * blockDim.x + threadIdx.x;
  long stride = (long)gridDim.x * blockDim.x;
  for (; i < n; i += stride) {
    int k = (int)(i / N);
    int c = (int)(i % N);
    out[(long)c * K + k] = (bf16)in[i];
  }
}

// ---------------------------------------------------------------------------
// Generic bf16 GEMM with fused epilogues.
//   C(M,N) = A(M,K) @ B(K,N); BT = B transposed (N x K row major, bf16).
//   Wave tile 16x64, 8 waves stacked in M -> block tile 128x64.
//   B tile (64 x 32k, 4KB) staged in LDS via TDM, double buffered.
// ---------------------------------------------------------------------------
struct EpiArgs {
  bf16* kdst;    // mode 0: K  dest  [b][h][ks][d]
  bf16* vtdst;   // mode 0: V^T dest [b][h][d][ks]
  bf16* qud;     // mode 1: Q+u dest [b][h][i][d]
  bf16* qvd;     // mode 1: Q+v dest [b][h][i][d]
  bf16* pdst;    // mode 2: P dest   [h][jp][d]
  float* fdst;   // mode 3: f32 dest [row][col]
  const float* ub;
  const float* vb;
};

#define GEMM_LDS_BYTES (2 * 64 * 32 * 2)   // double-buffered B tile

__global__ void __launch_bounds__(256) gemm_bf16(const bf16* __restrict__ A,
                                                 const bf16* __restrict__ BT,
                                                 int M, int N, int K, int mode,
                                                 EpiArgs ep) {
  extern __shared__ char smem[];
  bf16* btile = (bf16*)smem;                    // [2][64][32]

  const int lane  = threadIdx.x & 31;
  const int wave  = threadIdx.x >> 5;
  const int half  = lane >> 4;
  const int l16   = lane & 15;
  const int mrow0 = blockIdx.x * 128 + wave * 16;
  const int ncol0 = blockIdx.y * 64;

  v8f acc[4];
  for (int nt = 0; nt < 4; ++nt)
    for (int r = 0; r < 8; ++r) acc[nt][r] = 0.0f;

  const bf16* arow = A + (size_t)(mrow0 + l16) * K;
  const bf16* bbase = BT + (size_t)ncol0 * K;
  const int nk = K >> 5;

  if (wave == 0)
    tdm_load_2d(0, bbase, 32, 64, 32, 64, (unsigned long long)K);

  for (int kb = 0; kb < nk; ++kb) {
    const int kbase = kb * 32;
    const int buf = kb & 1;
    if (wave == 0) {
      if (kb + 1 < nk) {
        tdm_load_2d((unsigned)((buf ^ 1) * 64 * 32 * 2), bbase + kbase + 32,
                    32, 64, 32, 64, (unsigned long long)K);
        __builtin_amdgcn_s_wait_tensorcnt(1);
      } else {
        __builtin_amdgcn_s_wait_tensorcnt(0);
      }
    }
    __syncthreads();

    // A fragment: lane = row M, element e -> K=kbase+16*half+e
    v8bf alo = *(const v8bf*)(arow + kbase + half * 16);
    v8bf ahi = *(const v8bf*)(arow + kbase + half * 16 + 8);
    v16bf af;
    for (int e = 0; e < 8; ++e) { af[e] = alo[e]; af[e + 8] = ahi[e]; }
    __builtin_prefetch(arow + kbase + 64, 0, 0);   // global_prefetch_b8

    const bf16* bt = btile + buf * 64 * 32;
    for (int nt = 0; nt < 4; ++nt) {
      // B fragment from LDS: lane = col N, element e -> kk = 16*half + e
      v16bf bfr = *(const v16bf*)(bt + (size_t)(nt * 16 + l16) * 32 + half * 16);
      acc[nt] = __builtin_amdgcn_wmma_f32_16x16x32_bf16(
          false, af, false, bfr, (short)0, acc[nt], false, false);
    }
    __syncthreads();
  }

  // Epilogue: C layout — VGPR r holds M = r + 8*half, N = lane&15.
  for (int nt = 0; nt < 4; ++nt) {
    for (int r = 0; r < 8; ++r) {
      const int mrow = mrow0 + r + 8 * half;
      const int ncol = ncol0 + nt * 16 + l16;
      const float valf = acc[nt][r];
      if (mode == 0) {          // KV projection, rows = ks*BSZ + b
        const int ks = mrow / BSZ, b = mrow % BSZ;
        if (ncol < HD) {
          const int hh = ncol / DH, dd = ncol % DH;
          ep.kdst[(((size_t)b * NH + hh) * KSQ + ks) * DH + dd] = (bf16)valf;
        } else {
          const int c2 = ncol - HD;
          const int hh = c2 / DH, dd = c2 % DH;
          ep.vtdst[(((size_t)b * NH + hh) * DH + dd) * KSQ + ks] = (bf16)valf;
        }
      } else if (mode == 1) {   // Q projection + u/v bias, rows = i*BSZ + b
        const int iq = mrow / BSZ, b = mrow % BSZ;
        const int hh = ncol / DH, dd = ncol % DH;
        const size_t idx = (((size_t)b * NH + hh) * CS + iq) * DH + dd;
        ep.qud[idx] = (bf16)(valf + ep.ub[ncol]);
        ep.qvd[idx] = (bf16)(valf + ep.vb[ncol]);
      } else if (mode == 2) {   // P projection, rows = jp
        const int hh = ncol / DH, dd = ncol % DH;
        ep.pdst[((size_t)hh * KSQ + mrow) * DH + dd] = (bf16)valf;
      } else {                  // final projection -> f32 output
        ep.fdst[(size_t)mrow * N + ncol] = valf;
      }
    }
  }
}

// ---------------------------------------------------------------------------
// Fused TXL attention.
//   Block = 4 waves, 64 queries of one (b,h); each wave owns a 16-query tile.
//   Per 32-key chunk, wave 0 issues 3 TDM descriptors (K tile, V^T tile,
//   96-row P band shared across the 4 rel-shift windows), double buffered.
//   rel_shift: for unmasked (i,j): p-index = j + CS-1-i (no wraparound);
//   TDM zero-fills P rows >= KS via tensor_dim1.
// ---------------------------------------------------------------------------
#define KT_OFF   0                       // [2][32][64] bf16  (8 KB)
#define VT_OFF   (KT_OFF + 2 * 32 * 64 * 2)
#define PT_OFF   (VT_OFF + 2 * 64 * 32 * 2)   // [2][96][64] bf16 (24 KB)
#define SPL_OFF  (PT_OFF + 2 * 96 * 64 * 2)   // per-wave [16][48] f32 (12 KB)
#define PL_OFF   (SPL_OFF + 4 * 16 * 48 * 4)  // per-wave [16][32] bf16 (4 KB)
#define ATTN_LDS_BYTES (PL_OFF + 4 * 16 * 32 * 2)   // 56 KB

__global__ void __launch_bounds__(128) attn_kernel(const bf16* __restrict__ qu,
                                                   const bf16* __restrict__ qv,
                                                   const bf16* __restrict__ kbuf,
                                                   const bf16* __restrict__ vT,
                                                   const bf16* __restrict__ pbuf,
                                                   bf16* __restrict__ obuf) {
  extern __shared__ char smem[];

  const int lane  = threadIdx.x & 31;
  const int wave  = threadIdx.x >> 5;
  const int half  = lane >> 4;
  const int l16   = lane & 15;
  const int i0blk = blockIdx.x * 64;
  const int i0    = i0blk + wave * 16;      // this wave's query tile
  const int bh    = blockIdx.y;
  const int b     = bh / NH;
  const int h     = bh % NH;

  const bf16* kb  = kbuf + (size_t)bh * KSQ * DH;
  const bf16* vtb = vT   + (size_t)bh * DH * KSQ;
  const bf16* pb  = pbuf + (size_t)h  * KSQ * DH;

  float* spl = (float*)(smem + SPL_OFF) + wave * 16 * 48;
  bf16*  pl  = (bf16*)(smem + PL_OFF)  + wave * 16 * 32;

  // Qu/Qv A-fragments (D=64 -> two 32-k steps)
  const bf16* qurow = qu + ((size_t)bh * CS + (i0 + l16)) * DH;
  const bf16* qvrow = qv + ((size_t)bh * CS + (i0 + l16)) * DH;
  v16bf quf[2], qvf[2];
  for (int t = 0; t < 2; ++t) {
    const int k0 = t * 32 + half * 16;
    v8bf lo  = *(const v8bf*)(qurow + k0);
    v8bf hi  = *(const v8bf*)(qurow + k0 + 8);
    v8bf lo2 = *(const v8bf*)(qvrow + k0);
    v8bf hi2 = *(const v8bf*)(qvrow + k0 + 8);
    for (int e = 0; e < 8; ++e) {
      quf[t][e] = lo[e];  quf[t][e + 8] = hi[e];
      qvf[t][e] = lo2[e]; qvf[t][e + 8] = hi2[e];
    }
  }

  float mrun[8], lrun[8];
  v8f oacc[4];
  for (int r = 0; r < 8; ++r) { mrun[r] = -1e30f; lrun[r] = 0.0f; }
  for (int nt = 0; nt < 4; ++nt)
    for (int r = 0; r < 8; ++r) oacc[nt][r] = 0.0f;

  const int nchunk = (i0blk + 64 + PS + 31) / 32;

  // P-band union base for chunk ch: baseU = 32*ch + CS - 64 - i0blk.
  // Wave w's 48-row window starts at local row 48 - 16*w.
  auto issue_chunk = [&](int buf, int ch) {
    const int j0 = ch * 32;
    const int baseU = j0 + CS - 64 - i0blk;
    tdm_load_2d((unsigned)(KT_OFF + buf * 32 * 64 * 2), kb + (size_t)j0 * DH,
                64, 32, 64, 32, (unsigned long long)DH);
    tdm_load_2d((unsigned)(VT_OFF + buf * 64 * 32 * 2), vtb + j0,
                32, 64, 32, 64, (unsigned long long)KSQ);
    tdm_load_2d((unsigned)(PT_OFF + buf * 96 * 64 * 2), pb + (size_t)baseU * DH,
                64, 96, 64, (unsigned)(KSQ - baseU), (unsigned long long)DH);
  };

  if (wave == 0) issue_chunk(0, 0);

  for (int ch = 0; ch < nchunk; ++ch) {
    const int j0 = ch * 32;
    const int buf = ch & 1;
    if (wave == 0) {
      if (ch + 1 < nchunk) {
        issue_chunk(buf ^ 1, ch + 1);
        __builtin_amdgcn_s_wait_tensorcnt(3);
      } else {
        __builtin_amdgcn_s_wait_tensorcnt(0);
      }
    }
    __syncthreads();

    const bf16* Kt = (const bf16*)(smem + KT_OFF) + buf * 32 * 64;
    const bf16* Vt = (const bf16*)(smem + VT_OFF) + buf * 64 * 32;
    const bf16* Pt = (const bf16*)(smem + PT_OFF) + buf * 96 * 64;

    // ---- content scores: two 16x16 tiles from LDS K tile
    v8f sc[2];
    for (int s = 0; s < 2; ++s) for (int r = 0; r < 8; ++r) sc[s][r] = 0.0f;
    for (int s = 0; s < 2; ++s) {
      for (int t = 0; t < 2; ++t) {
        v16bf kf = *(const v16bf*)(Kt + (size_t)(s * 16 + l16) * DH +
                                   t * 32 + half * 16);
        sc[s] = __builtin_amdgcn_wmma_f32_16x16x32_bf16(
            false, quf[t], false, kf, (short)0, sc[s], false, false);
      }
    }

    // ---- positional scores: wave's 48-row window inside the 96-row band
    const int prow0 = 48 - 16 * wave;
    for (int pt = 0; pt < 3; ++pt) {
      v8f sp;
      for (int r = 0; r < 8; ++r) sp[r] = 0.0f;
      const int prow = prow0 + pt * 16 + l16;
      for (int t = 0; t < 2; ++t) {
        v16bf pf = *(const v16bf*)(Pt + (size_t)prow * DH + t * 32 + half * 16);
        sp = __builtin_amdgcn_wmma_f32_16x16x32_bf16(
            false, qvf[t], false, pf, (short)0, sp, false, false);
      }
      for (int r = 0; r < 8; ++r)
        spl[(r + 8 * half) * 48 + pt * 16 + l16] = sp[r];
    }
    __syncthreads();

    // ---- combine with rel_shift, mask, online softmax
    float sarr[2][8];
    for (int r = 0; r < 8; ++r) {
      const int m = r + 8 * half;
      const int iq = i0 + m;
      for (int s = 0; s < 2; ++s) {
        const int jcol = j0 + s * 16 + l16;
        const int c = s * 16 + l16 + 15 - m;        // window-local, in [0,46]
        const float sval = (sc[s][r] + spl[m * 48 + c]) * 0.125f;  // 1/sqrt(64)
        sarr[s][r] = (jcol <= iq + PS) ? sval : -1e30f;
      }
    }
    for (int r = 0; r < 8; ++r) {
      float rm = fmaxf(sarr[0][r], sarr[1][r]);
      for (int off = 1; off < 16; off <<= 1)
        rm = fmaxf(rm, __shfl_xor(rm, off, 32));
      const float newm = fmaxf(mrun[r], rm);
      const float alpha = __expf(mrun[r] - newm);
      mrun[r] = newm;
      const float p0 = __expf(sarr[0][r] - newm);
      const float p1 = __expf(sarr[1][r] - newm);
      float ls = p0 + p1;
      for (int off = 1; off < 16; off <<= 1) ls += __shfl_xor(ls, off, 32);
      lrun[r] = lrun[r] * alpha + ls;
      for (int nt = 0; nt < 4; ++nt) oacc[nt][r] *= alpha;
      const int m = r + 8 * half;
      pl[m * 32 + l16]      = (bf16)p0;
      pl[m * 32 + 16 + l16] = (bf16)p1;
    }
    __syncthreads();

    // ---- O += probs(16x32) @ V(32x64), V^T fragments from LDS
    v16bf af;
    {
      v8bf lo = *(const v8bf*)(pl + l16 * 32 + half * 16);
      v8bf hi = *(const v8bf*)(pl + l16 * 32 + 8 + half * 16);
      for (int e = 0; e < 8; ++e) { af[e] = lo[e]; af[e + 8] = hi[e]; }
    }
    for (int nt = 0; nt < 4; ++nt) {
      v16bf vf = *(const v16bf*)(Vt + (size_t)(nt * 16 + l16) * 32 + half * 16);
      oacc[nt] = __builtin_amdgcn_wmma_f32_16x16x32_bf16(
          false, af, false, vf, (short)0, oacc[nt], false, false);
    }
    __syncthreads();
  }

  // ---- normalize and write O (rows = i*BSZ+b, cols = h*DH+d) as bf16
  for (int nt = 0; nt < 4; ++nt) {
    for (int r = 0; r < 8; ++r) {
      const int m = r + 8 * half;
      const int iq = i0 + m;
      const float val = oacc[nt][r] / lrun[r];
      const int col = h * DH + nt * 16 + l16;
      obuf[((size_t)iq * BSZ + b) * HD + col] = (bf16)val;
    }
  }
}

// ---------------------------------------------------------------------------
// Host orchestration
// ---------------------------------------------------------------------------
extern "C" void kernel_launch(void* const* d_in, const int* in_sizes, int n_in,
                              void* d_out, int out_size, void* d_ws, size_t ws_size,
                              hipStream_t stream) {
  const float* input_ = (const float*)d_in[0];
  const float* pos    = (const float*)d_in[1];
  const float* memory = (const float*)d_in[2];
  const float* u_b    = (const float*)d_in[3];
  const float* v_b    = (const float*)d_in[4];
  const float* W_kv   = (const float*)d_in[5];
  const float* W_q    = (const float*)d_in[6];
  const float* W_p    = (const float*)d_in[7];
  const float* W_out  = (const float*)d_in[8];
  // d_in[9] (mask) is a pure index function; recomputed in-kernel.
  (void)in_sizes; (void)n_in; (void)out_size; (void)ws_size;

  char* ws = (char*)d_ws;
  size_t off = 0;
  auto alloc_bf = [&](size_t elems) -> bf16* {
    bf16* p = (bf16*)(ws + off);
    off += elems * sizeof(bf16);
    off = (off + 255) & ~(size_t)255;
    return p;
  };

  bf16* xm    = alloc_bf((size_t)KSQ * BSZ * DIN);   // concat(memory, input) bf16
  bf16* posb  = alloc_bf((size_t)KSQ * DIN);
  bf16* wkvT  = alloc_bf((size_t)2 * HD * DIN);
  bf16* wqT   = alloc_bf((size_t)HD * DIN);
  bf16* wpT   = alloc_bf((size_t)HD * DIN);
  bf16* woutT = alloc_bf((size_t)DIN * HD);
  bf16* kbuf  = alloc_bf((size_t)BSZ * NH * KSQ * DH);
  bf16* vTb   = alloc_bf((size_t)BSZ * NH * DH * KSQ);
  bf16* qub   = alloc_bf((size_t)BSZ * NH * CS * DH);
  bf16* qvb   = alloc_bf((size_t)BSZ * NH * CS * DH);
  bf16* pbuf  = alloc_bf((size_t)NH * KSQ * DH);
  bf16* obuf  = alloc_bf((size_t)CS * BSZ * HD);

  const dim3 cvt_grid(2048), cvt_blk(256);
  cvt_copy<<<cvt_grid, cvt_blk, 0, stream>>>(memory, xm, (long)PS * BSZ * DIN);
  cvt_copy<<<cvt_grid, cvt_blk, 0, stream>>>(input_, xm + (size_t)PS * BSZ * DIN,
                                             (long)CS * BSZ * DIN);
  cvt_copy<<<cvt_grid, cvt_blk, 0, stream>>>(pos, posb, (long)KSQ * DIN);
  cvt_transpose<<<cvt_grid, cvt_blk, 0, stream>>>(W_kv, wkvT, DIN, 2 * HD);
  cvt_transpose<<<cvt_grid, cvt_blk, 0, stream>>>(W_q, wqT, DIN, HD);
  cvt_transpose<<<cvt_grid, cvt_blk, 0, stream>>>(W_p, wpT, DIN, HD);
  cvt_transpose<<<cvt_grid, cvt_blk, 0, stream>>>(W_out, woutT, HD, DIN);

  EpiArgs ep{};
  ep.ub = u_b; ep.vb = v_b;
  ep.kdst = kbuf; ep.vtdst = vTb; ep.qud = qub; ep.qvd = qvb;
  ep.pdst = pbuf; ep.fdst = (float*)d_out;

  // KV projection: (KS*BSZ x DIN) @ (DIN x 2HD)
  gemm_bf16<<<dim3((KSQ * BSZ) / 128, (2 * HD) / 64), 256, GEMM_LDS_BYTES,
              stream>>>(xm, wkvT, KSQ * BSZ, 2 * HD, DIN, 0, ep);
  // Q projection (+u/+v): (CS*BSZ x DIN) @ (DIN x HD)
  gemm_bf16<<<dim3((CS * BSZ) / 128, HD / 64), 256, GEMM_LDS_BYTES, stream>>>(
      xm + (size_t)PS * BSZ * DIN, wqT, CS * BSZ, HD, DIN, 1, ep);
  // P projection: (KS x DIN) @ (DIN x HD)
  gemm_bf16<<<dim3(KSQ / 128, HD / 64), 256, GEMM_LDS_BYTES, stream>>>(
      posb, wpT, KSQ, HD, DIN, 2, ep);

  // Fused attention: 64-query blocks x (b,h), 4 waves per block
  attn_kernel<<<dim3(CS / 64, BSZ * NH), 128, ATTN_LDS_BYTES, stream>>>(
      qub, qvb, kbuf, vTb, pbuf, obuf);

  // Output projection: (CS*BSZ x HD) @ (HD x DIN) -> f32 d_out
  gemm_bf16<<<dim3((CS * BSZ) / 128, DIN / 64), 256, GEMM_LDS_BYTES, stream>>>(
      obuf, woutT, CS * BSZ, DIN, HD, 3, ep);
}
